// RCCAModule_11931419148311
// MI455X (gfx1250) — matrix operand order repaired and checked by reading, not compiled, not run on hardware
//
#include <hip/hip_runtime.h>
#include <cstddef>
#include <cstdint>

typedef unsigned short u16;
typedef unsigned int   u32;
typedef __attribute__((ext_vector_type(16))) __bf16 v16bf;
typedef __attribute__((ext_vector_type(8)))  float  v8f;

#define BB    2
#define HH    97
#define WWI   97
#define HWP   9409      // 97*97
#define PP    18818     // 2*97*97
#define CIN   2048
#define CMID  512
#define CQ    64
#define NCLS  19

static __device__ __forceinline__ u16 f2bf(float f) {
  u32 u = __float_as_uint(f);
  u32 r = (u + 0x7FFFu + ((u >> 16) & 1u)) >> 16;
  return (u16)r;
}
static __device__ __forceinline__ float bf2f(u16 h) {
  return __uint_as_float(((u32)h) << 16);
}

union FragU { uint4 q[2]; v16bf v; };

// ---------------------------------------------------------------------------
// NCHW f32 -> NHWC bf16 cast/transpose
// ---------------------------------------------------------------------------
__global__ __launch_bounds__(256) void k_cast_nchw_nhwc(
    const float* __restrict__ x, u16* __restrict__ o, int C) {
  long tot = (long)BB * C * HWP;
  for (long idx = (long)blockIdx.x * blockDim.x + threadIdx.x; idx < tot;
       idx += (long)gridDim.x * blockDim.x) {
    long b = idx / ((long)C * HWP);
    long r = idx - b * (long)C * HWP;
    int  c = (int)(r / HWP);
    int hw = (int)(r - (long)c * HWP);
    o[(b * HWP + hw) * (long)C + c] = f2bf(x[idx]);
  }
}

// ---------------------------------------------------------------------------
// Pack conv weight OIHW (or OC for 1x1) into WMMA-friendly tiles:
// k = (dh*3+dw)*Cin + c ; Wt[(k>>5)][o][k&31], o padded to CoutPad (zero fill)
// ---------------------------------------------------------------------------
__global__ __launch_bounds__(256) void k_pack_w(
    const float* __restrict__ w, u16* __restrict__ wt,
    int Cout, int CoutPad, int Cin, int khw) {
  long K   = (long)Cin * khw;
  long tot = K * CoutPad;
  for (long idx = (long)blockIdx.x * blockDim.x + threadIdx.x; idx < tot;
       idx += (long)gridDim.x * blockDim.x) {
    long k = idx / CoutPad;
    int  o = (int)(idx - k * CoutPad);
    float v = 0.f;
    if (o < Cout) {
      int sp = (int)(k / Cin);
      int c  = (int)(k - (long)sp * Cin);
      v = w[((long)o * Cin + c) * khw + sp];
    }
    wt[(k >> 5) * ((long)CoutPad << 5) + ((long)o << 5) + (k & 31)] = f2bf(v);
  }
}

// ---------------------------------------------------------------------------
// BN folding: inv = s*rsqrt(v+eps); beta = b - m*inv
// ---------------------------------------------------------------------------
__global__ void k_bn_prep(const float* s, const float* bi, const float* m,
                          const float* v, float* inv, float* bet, int C) {
  int i = blockIdx.x * blockDim.x + threadIdx.x;
  if (i < C) {
    float iv = s[i] * rsqrtf(v[i] + 1e-5f);
    inv[i] = iv;
    bet[i] = bi[i] - m[i] * iv;
  }
}

// ---------------------------------------------------------------------------
// Implicit-GEMM conv (3x3 via khw=9, 1x1 via khw=1), NHWC bf16 activations.
// Tile: BM=256 pixels x BN=64 outputs x BK=32, 256 threads / 8 waves.
// Each wave: 32x64 tile = 2x4 v_wmma_f32_16x16x32_bf16 per K-step.
// Spatial decode hoisted out of the channel loop (no inner-loop division).
// Weight tile staged LDS-direct via global_load_async_to_lds_b128 (ASYNCcnt).
// Two-source A gather (A0 up to csplit channels, then A1) for concat input.
// Epilogue: y = acc*inv[o] + beta[o] (+ReLU); writes NHWC bf16 and/or NCHW f32.
// ---------------------------------------------------------------------------
__global__ __launch_bounds__(256) void k_conv_gemm(
    const u16* __restrict__ A0, const u16* __restrict__ A1, int csplit,
    const u16* __restrict__ Wt,
    const float* __restrict__ bnInv, const float* __restrict__ bnBeta,
    u16* __restrict__ outBf, float* __restrict__ outF32,
    int Cin, int Cout, int CoutPad, int khw, int relu) {
  __shared__ __align__(16) u16 As[256 * 48];  // [pixel][k], pitch 48 (96B rows)
  __shared__ __align__(16) u16 Bs[64 * 48];   // [o][k], pitch 48

  const int t     = threadIdx.x;
  const int mbase = blockIdx.x * 256;
  const int nbase = blockIdx.y * 64;

  const int olB = t >> 2, kbB = (t & 3) << 3;   // B tile: 64x32, 8 elem/thread
  const int wid = t >> 5, lane = t & 31;
  const int mw  = wid << 5;                     // 32 M-rows per wave
  const int lm  = lane & 15, lh = lane >> 4;
  const int a1s = Cin - csplit;

  const v8f z8 = {0.f, 0.f, 0.f, 0.f, 0.f, 0.f, 0.f, 0.f};
  v8f acc[2][4];
#pragma unroll
  for (int i = 0; i < 2; ++i)
#pragma unroll
    for (int j = 0; j < 4; ++j) acc[i][j] = z8;

  // One pixel row per thread: decode once.
  const int gp = mbase + t;
  int pb = 0, ph = 0, pw = 0;
  const bool prow = gp < PP;
  if (prow) {
    pb = gp / HWP;
    int hw = gp - pb * HWP;
    ph = hw / WWI;
    pw = hw - ph * WWI;
  }

  // Per-lane LDS destination for the async weight tile (LDS offset = low 32 bits
  // of the generic shared-space pointer).
  const u32 ldsB = (u32)(size_t)(const void*)&Bs[olB * 48 + kbB];
  u16* const asRow = &As[t * 48];

  int ko = 0;
  for (int sp = 0; sp < khw; ++sp) {
    int dh = 0, dw = 0;
    if (khw == 9) {
      dh = sp / 3 - 1;
      dw = sp - (sp / 3) * 3 - 1;
    }
    // Hoisted spatial validity + row base pointers for this (dh,dw).
    bool rv = false;
    const u16* r0 = A0;
    const u16* r1 = A1;
    if (prow) {
      int y = ph + dh, xx = pw + dw;
      if ((unsigned)y < (unsigned)HH && (unsigned)xx < (unsigned)WWI) {
        long row = (long)(pb * HH + y) * WWI + xx;
        r0 = A0 + row * (long)csplit;
        if (A1) r1 = A1 + row * (long)a1s;
        rv = true;
      }
    }

    for (int cb = 0; cb < Cin; cb += 32, ++ko) {
      // Gather this thread's 32-channel A row chunk (zero padded).
      uint4 av0 = make_uint4(0, 0, 0, 0), av1 = av0, av2 = av0, av3 = av0;
      if (rv) {
        const u16* src = (cb < csplit) ? (r0 + cb) : (r1 + (cb - csplit));
        const uint4* s4 = (const uint4*)src;
        av0 = s4[0]; av1 = s4[1]; av2 = s4[2]; av3 = s4[3];
        __builtin_prefetch(src + 32, 0, 0);   // global_prefetch_b8 (next chunk)
      }

      __syncthreads();   // previous K-step's LDS reads done

      // Async DMA of weight tile straight into LDS (tracked by ASYNCcnt).
      {
        unsigned long long ga = (unsigned long long)(size_t)(const void*)
            (Wt + (((long)ko * CoutPad) + (nbase + olB)) * 32 + kbB);
        asm volatile("global_load_async_to_lds_b128 %0, %1, off"
                     :: "v"(ldsB), "v"(ga) : "memory");
      }

      *(uint4*)(asRow + 0)  = av0;
      *(uint4*)(asRow + 8)  = av1;
      *(uint4*)(asRow + 16) = av2;
      *(uint4*)(asRow + 24) = av3;

      asm volatile("s_wait_asynccnt 0x0" ::: "memory");
      __syncthreads();   // A + async-B tiles visible to all waves

      FragU af[2];
#pragma unroll
      for (int mi = 0; mi < 2; ++mi) {   // A 16x32: lane m=L&15, K runs lh*8, 16+lh*8
        const u16* ar = &As[(mw + mi * 16 + lm) * 48];
        af[mi].q[0] = *(const uint4*)(ar + lh * 8);
        af[mi].q[1] = *(const uint4*)(ar + 16 + lh * 8);
      }
#pragma unroll
      for (int ni = 0; ni < 4; ++ni) {   // B 32x16: lane n=L&15, K runs lh*8, 16+lh*8
        FragU bfr;
        const u16* br = &Bs[(ni * 16 + lm) * 48];
        bfr.q[0] = *(const uint4*)(br + lh * 8);
        bfr.q[1] = *(const uint4*)(br + 16 + lh * 8);
        acc[0][ni] = __builtin_amdgcn_wmma_f32_16x16x32_bf16(
            false, af[0].v, false, bfr.v, (short)0, acc[0][ni], false, false);
        acc[1][ni] = __builtin_amdgcn_wmma_f32_16x16x32_bf16(
            false, af[1].v, false, bfr.v, (short)0, acc[1][ni], false, false);
      }
    }
  }

#pragma unroll
  for (int mi = 0; mi < 2; ++mi) {
#pragma unroll
    for (int ni = 0; ni < 4; ++ni) {
      int nn = nbase + ni * 16 + lm;
      bool nok = nn < Cout;
      float inv = 1.f, bet = 0.f;
      if (nok) {
        if (bnInv)  inv = bnInv[nn];
        if (bnBeta) bet = bnBeta[nn];
      }
#pragma unroll
      for (int e = 0; e < 8; ++e) {      // C/D: VGPR e -> M = e + lh*8
        int mm = mbase + mw + mi * 16 + e + lh * 8;
        if (mm < PP && nok) {
          float v = acc[mi][ni][e] * inv + bet;
          if (relu) v = fmaxf(v, 0.f);
          if (outBf) outBf[(long)mm * Cout + nn] = f2bf(v);
          if (outF32) {
            int bb = mm / HWP, hw = mm - bb * HWP;
            outF32[((long)(bb * Cout + nn)) * HWP + hw] = v;
          }
        }
      }
    }
  }
}

// ---------------------------------------------------------------------------
// Criss-cross energies. Per line (b,h) [dir=0] or (b,w) [dir=1]:
//   E(j,w)=sum_c K(j,c)Q(w,c)  (112x112x64 WMMA, pad 97->112), 224 thr / 7 waves
// Col part gets -inf on the diagonal. E layout: [b][194][h][w] f32.
// ---------------------------------------------------------------------------
__global__ __launch_bounds__(256) void k_energy(
    const u16* __restrict__ qb, const u16* __restrict__ kb, float* __restrict__ E) {
  __shared__ __align__(16) u16 Kt[112 * 80];
  __shared__ __align__(16) u16 Qt[112 * 80];
  const int t   = threadIdx.x;   // 224
  const int dir = blockIdx.y;
  const int b   = blockIdx.x / HH;
  const int li  = blockIdx.x - b * HH;

  {
    int r  = t >> 1;             // 0..111 (row of K/Q matrices)
    int cb = (t & 1) << 5;       // 0 or 32
    uint4 kz[4], qz[4];
#pragma unroll
    for (int i = 0; i < 4; ++i) { kz[i] = make_uint4(0, 0, 0, 0); qz[i] = kz[i]; }
    if (r < HH) {
      long roff = (dir == 0) ? (((long)(b * HH + li) * WWI + r) * CQ + cb)
                             : (((long)(b * HH + r) * WWI + li) * CQ + cb);
      const uint4* ks = (const uint4*)(kb + roff);
      const uint4* qs = (const uint4*)(qb + roff);
#pragma unroll
      for (int i = 0; i < 4; ++i) { kz[i] = ks[i]; qz[i] = qs[i]; }
    }
#pragma unroll
    for (int i = 0; i < 4; ++i) {
      *(uint4*)(&Kt[r * 80 + cb + i * 8]) = kz[i];
      *(uint4*)(&Qt[r * 80 + cb + i * 8]) = qz[i];
    }
  }
  __syncthreads();

  const int wv = t >> 5;                 // 0..6 -> 16-row M strip
  const int lane = t & 31, lm = lane & 15, lh = lane >> 4;
  const int m = wv << 4;
  const v8f z8 = {0.f, 0.f, 0.f, 0.f, 0.f, 0.f, 0.f, 0.f};

  FragU af[2];
#pragma unroll
  for (int cc = 0; cc < 2; ++cc) {
    const u16* ar = &Kt[(m + lm) * 80 + cc * 32];
    af[cc].q[0] = *(const uint4*)(ar + lh * 8);
    af[cc].q[1] = *(const uint4*)(ar + 16 + lh * 8);
  }

  for (int n = 0; n < 7; ++n) {
    v8f acc = z8;
#pragma unroll
    for (int cc = 0; cc < 2; ++cc) {
      FragU bfv;
      const u16* br = &Qt[((n << 4) + lm) * 80 + cc * 32];
      bfv.q[0] = *(const uint4*)(br + lh * 8);
      bfv.q[1] = *(const uint4*)(br + 16 + lh * 8);
      acc = __builtin_amdgcn_wmma_f32_16x16x32_bf16(
          false, af[cc].v, false, bfv.v, (short)0, acc, false, false);
    }
#pragma unroll
    for (int e = 0; e < 8; ++e) {
      int Mj = m + e + lh * 8;
      int Nw = (n << 4) + lm;
      if (Mj < HH && Nw < WWI) {
        if (dir == 0) {
          E[(((long)(b * 194 + Mj)) * HH + li) * WWI + Nw] = acc[e];
        } else {
          float v = acc[e];
          if (Mj == Nw) v = -__builtin_inff();
          E[(((long)(b * 194 + 97 + Mj)) * HH + Nw) * WWI + li] = v;
        }
      }
    }
  }
}

// ---------------------------------------------------------------------------
// Softmax over the 194 concat channels per pixel; outputs bf16 attention.
// ---------------------------------------------------------------------------
__global__ __launch_bounds__(256) void k_softmax(
    const float* __restrict__ E, u16* __restrict__ attnb) {
  int p = blockIdx.x * 256 + threadIdx.x;
  if (p >= PP) return;
  int b = p / HWP, hw = p - b * HWP;
  const float* e = E + (long)b * 194 * HWP + hw;
  float mx = -3.4e38f;
  for (int ch = 0; ch < 194; ++ch) mx = fmaxf(mx, e[(long)ch * HWP]);
  float s = 0.f;
  for (int ch = 0; ch < 194; ++ch) s += __expf(e[(long)ch * HWP] - mx);
  float inv = 1.f / s;
  u16* o = attnb + (long)b * 194 * HWP + hw;
  for (int ch = 0; ch < 194; ++ch)
    o[(long)ch * HWP] = f2bf(__expf(e[(long)ch * HWP] - mx) * inv);
}

// ---------------------------------------------------------------------------
// Row aggregation. Per (b,h,cblk64): Out(w,c) = sum_j A(j,w) V(j,c).
// A^T staged [w][j] (K padded to 128), V^T staged [c][j]. 256 thr / 7 compute waves.
// ---------------------------------------------------------------------------
__global__ __launch_bounds__(256) void k_agg_row(
    const u16* __restrict__ attnb, const u16* __restrict__ vb,
    float* __restrict__ oatt) {
  __shared__ __align__(16) u16 At[112 * 136];
  __shared__ __align__(16) u16 Vt[64 * 40];
  const int t = threadIdx.x;
  const int b = blockIdx.x / HH, li = blockIdx.x - b * HH;  // li = h
  const int cblk = blockIdx.y * 64;

  {
    int j  = t >> 1;                 // 0..127 (K incl. pad)
    int wb = (t & 1) * 56;
    bool jok = j < HH;
    const u16* src = attnb + (((long)(b * 194 + j)) * HH + li) * WWI;
    for (int i = 0; i < 56; ++i) {
      int w = wb + i;
      u16 v = 0;
      if (jok && w < WWI) v = src[w];
      At[w * 136 + j] = v;
    }
  }

  const int wv = t >> 5, lane = t & 31, lm = lane & 15, lh = lane >> 4;
  const int m = wv << 4;
  const v8f z8 = {0.f, 0.f, 0.f, 0.f, 0.f, 0.f, 0.f, 0.f};
  v8f acc[4];
  acc[0] = z8; acc[1] = z8; acc[2] = z8; acc[3] = z8;

  for (int jb = 0; jb < 4; ++jb) {
    __syncthreads();
    {
      int j  = t >> 3;               // 0..31
      int c8 = (t & 7) << 3;
      int ja = (jb << 5) + j;
      uint4 vv = make_uint4(0, 0, 0, 0);
      if (ja < HH)
        vv = *(const uint4*)(vb + (((long)(b * HH + li)) * WWI + ja) * (long)CMID + cblk + c8);
      const u16* pv = (const u16*)&vv;
#pragma unroll
      for (int e = 0; e < 8; ++e) Vt[(c8 + e) * 40 + j] = pv[e];
    }
    __syncthreads();
    if (wv < 7) {
      FragU af;
      const u16* ar = &At[(m + lm) * 136 + (jb << 5)];
      af.q[0] = *(const uint4*)(ar + lh * 8);
      af.q[1] = *(const uint4*)(ar + 16 + lh * 8);
#pragma unroll
      for (int n = 0; n < 4; ++n) {
        FragU bfv;
        const u16* br = &Vt[((n << 4) + lm) * 40];
        bfv.q[0] = *(const uint4*)(br + lh * 8);
        bfv.q[1] = *(const uint4*)(br + 16 + lh * 8);
        acc[n] = __builtin_amdgcn_wmma_f32_16x16x32_bf16(
            false, af.v, false, bfv.v, (short)0, acc[n], false, false);
      }
    }
  }
  if (wv < 7) {
#pragma unroll
    for (int n = 0; n < 4; ++n)
#pragma unroll
      for (int e = 0; e < 8; ++e) {
        int w = m + e + lh * 8;
        int c = cblk + (n << 4) + lm;
        if (w < WWI)
          oatt[(((long)(b * HH + li)) * WWI + w) * (long)CMID + c] = acc[n][e];
      }
  }
}

// ---------------------------------------------------------------------------
// Column aggregation + combine: out = gamma*(row+col) + residual, bf16 NHWC.
// Per (b,w,cblk64): Out(h,c) = sum_i A(i,h) V(i,c).
// ---------------------------------------------------------------------------
__global__ __launch_bounds__(256) void k_agg_col(
    const u16* __restrict__ attnb, const u16* __restrict__ vb,
    const float* __restrict__ oatt, const u16* __restrict__ xprev,
    const float* __restrict__ gamma, u16* __restrict__ outb) {
  __shared__ __align__(16) u16 At[112 * 136];
  __shared__ __align__(16) u16 Vt[64 * 40];
  const int t = threadIdx.x;
  const int b = blockIdx.x / HH, li = blockIdx.x - b * HH;  // li = w
  const int cblk = blockIdx.y * 64;

  {
    int i  = t >> 1;                 // 0..127 (K incl. pad)
    int hb = (t & 1) * 56;
    bool iok = i < HH;
    const u16* src = attnb + ((long)(b * 194 + 97 + i)) * HWP + li;
    for (int q = 0; q < 56; ++q) {
      int h = hb + q;
      u16 v = 0;
      if (iok && h < HH) v = src[(long)h * WWI];
      At[h * 136 + i] = v;
    }
  }

  const int wv = t >> 5, lane = t & 31, lm = lane & 15, lh = lane >> 4;
  const int m = wv << 4;
  const v8f z8 = {0.f, 0.f, 0.f, 0.f, 0.f, 0.f, 0.f, 0.f};
  v8f acc[4];
  acc[0] = z8; acc[1] = z8; acc[2] = z8; acc[3] = z8;

  for (int jb = 0; jb < 4; ++jb) {
    __syncthreads();
    {
      int j  = t >> 3;
      int c8 = (t & 7) << 3;
      int ia = (jb << 5) + j;
      uint4 vv = make_uint4(0, 0, 0, 0);
      if (ia < HH)
        vv = *(const uint4*)(vb + (((long)(b * HH + ia)) * WWI + li) * (long)CMID + cblk + c8);
      const u16* pv = (const u16*)&vv;
#pragma unroll
      for (int e = 0; e < 8; ++e) Vt[(c8 + e) * 40 + j] = pv[e];
    }
    __syncthreads();
    if (wv < 7) {
      FragU af;
      const u16* ar = &At[(m + lm) * 136 + (jb << 5)];
      af.q[0] = *(const uint4*)(ar + lh * 8);
      af.q[1] = *(const uint4*)(ar + 16 + lh * 8);
#pragma unroll
      for (int n = 0; n < 4; ++n) {
        FragU bfv;
        const u16* br = &Vt[((n << 4) + lm) * 40];
        bfv.q[0] = *(const uint4*)(br + lh * 8);
        bfv.q[1] = *(const uint4*)(br + 16 + lh * 8);
        acc[n] = __builtin_amdgcn_wmma_f32_16x16x32_bf16(
            false, af.v, false, bfv.v, (short)0, acc[n], false, false);
      }
    }
  }
  if (wv < 7) {
    float g = gamma[0];
#pragma unroll
    for (int n = 0; n < 4; ++n)
#pragma unroll
      for (int e = 0; e < 8; ++e) {
        int h = m + e + lh * 8;
        int c = cblk + (n << 4) + lm;
        if (h < HH) {
          long pi = ((long)(b * HH + h)) * WWI + li;
          float rowp = oatt[pi * CMID + c];
          float val = g * (rowp + acc[n][e]) + bf2f(xprev[pi * CMID + c]);
          outb[pi * CMID + c] = f2bf(val);
        }
      }
  }
}

// ---------------------------------------------------------------------------
// Orchestration
// ---------------------------------------------------------------------------
extern "C" void kernel_launch(void* const* d_in, const int* in_sizes, int n_in,
                              void* d_out, int out_size, void* d_ws, size_t ws_size,
                              hipStream_t stream) {
  const float* x       = (const float*)d_in[0];
  const float* conva_w = (const float*)d_in[1];
  const float* bn1_s   = (const float*)d_in[2];
  const float* bn1_b   = (const float*)d_in[3];
  const float* bn1_m   = (const float*)d_in[4];
  const float* bn1_v   = (const float*)d_in[5];
  const float* q_w     = (const float*)d_in[6];
  const float* q_b     = (const float*)d_in[7];
  const float* k_w     = (const float*)d_in[8];
  const float* k_b     = (const float*)d_in[9];
  const float* v_w     = (const float*)d_in[10];
  const float* v_b     = (const float*)d_in[11];
  const float* gamma   = (const float*)d_in[12];
  const float* convb_w = (const float*)d_in[13];
  const float* bn2_s   = (const float*)d_in[14];
  const float* bn2_b   = (const float*)d_in[15];
  const float* bn2_m   = (const float*)d_in[16];
  const float* bn2_v   = (const float*)d_in[17];
  const float* bott_w  = (const float*)d_in[18];
  const float* bn3_s   = (const float*)d_in[19];
  const float* bn3_b   = (const float*)d_in[20];
  const float* bn3_m   = (const float*)d_in[21];
  const float* bn3_v   = (const float*)d_in[22];
  const float* cls_w   = (const float*)d_in[23];
  const float* cls_b   = (const float*)d_in[24];
  // d_in[25] = recurrence: fixed at 2 (cannot sync-read device scalar under capture)

  size_t off = 0;
  auto alloc = [&](size_t bytes) -> void* {
    void* p = (char*)d_ws + off;
    off += (bytes + 255) & ~(size_t)255;
    return p;
  };

  u16* x_nhwc = (u16*)alloc((size_t)PP * CIN * 2);
  u16* wA     = (u16*)alloc((size_t)CIN * 9 * CMID * 2);
  u16* wB     = (u16*)alloc((size_t)CMID * 9 * CMID * 2);
  u16* wBt    = (u16*)alloc((size_t)(CIN + CMID) * 9 * CMID * 2);
  u16* wQ     = (u16*)alloc((size_t)CMID * 64 * 2);
  u16* wK     = (u16*)alloc((size_t)CMID * 64 * 2);
  u16* wV     = (u16*)alloc((size_t)CMID * CMID * 2);
  u16* wCls   = (u16*)alloc((size_t)CMID * 64 * 2);
  float* bn1i = (float*)alloc(CMID * 4); float* bn1bt = (float*)alloc(CMID * 4);
  float* bn2i = (float*)alloc(CMID * 4); float* bn2bt = (float*)alloc(CMID * 4);
  float* bn3i = (float*)alloc(CMID * 4); float* bn3bt = (float*)alloc(CMID * 4);
  u16* act1 = (u16*)alloc((size_t)PP * CMID * 2);
  u16* res1 = (u16*)alloc((size_t)PP * CMID * 2);
  u16* res2 = (u16*)alloc((size_t)PP * CMID * 2);
  u16* qb   = (u16*)alloc((size_t)PP * CQ * 2);
  u16* kb   = (u16*)alloc((size_t)PP * CQ * 2);
  u16* vbuf = (u16*)alloc((size_t)PP * CMID * 2);
  float* E  = (float*)alloc((size_t)BB * 194 * HWP * 4);
  u16* attn = (u16*)alloc((size_t)BB * 194 * HWP * 2);
  float* oatt = (float*)alloc((size_t)PP * CMID * 4);
  u16* act2 = (u16*)alloc((size_t)PP * CMID * 2);
  u16* ybuf = (u16*)alloc((size_t)PP * CMID * 2);
  (void)ws_size; (void)in_sizes; (void)n_in; (void)out_size;

  // Input layout conversion
  {
    long tot = (long)BB * CIN * HWP;
    int g = (int)((tot + 255) / 256);
    k_cast_nchw_nhwc<<<g, 256, 0, stream>>>(x, x_nhwc, CIN);
  }
  // Weight packing
  auto packLaunch = [&](const float* w, u16* wt, int Cout, int CoutPad, int Cin, int khw) {
    long tot = (long)Cin * khw * CoutPad;
    int g = (int)((tot + 255) / 256);
    if (g > 65535) g = 65535;
    k_pack_w<<<g, 256, 0, stream>>>(w, wt, Cout, CoutPad, Cin, khw);
  };
  packLaunch(conva_w, wA, CMID, CMID, CIN, 9);
  packLaunch(convb_w, wB, CMID, CMID, CMID, 9);
  packLaunch(bott_w, wBt, CMID, CMID, CIN + CMID, 9);
  packLaunch(q_w, wQ, CQ, 64, CMID, 1);
  packLaunch(k_w, wK, CQ, 64, CMID, 1);
  packLaunch(v_w, wV, CMID, CMID, CMID, 1);
  packLaunch(cls_w, wCls, NCLS, 64, CMID, 1);
  // BN folding
  k_bn_prep<<<2, 256, 0, stream>>>(bn1_s, bn1_b, bn1_m, bn1_v, bn1i, bn1bt, CMID);
  k_bn_prep<<<2, 256, 0, stream>>>(bn2_s, bn2_b, bn2_m, bn2_v, bn2i, bn2bt, CMID);
  k_bn_prep<<<2, 256, 0, stream>>>(bn3_s, bn3_b, bn3_m, bn3_v, bn3i, bn3bt, CMID);

  const int MT = (PP + 255) / 256;  // 74 M-tiles of 256 pixels

  // conva + bn1 + relu
  k_conv_gemm<<<dim3(MT, 8), 256, 0, stream>>>(
      x_nhwc, nullptr, CIN, wA, bn1i, bn1bt, act1, nullptr, CIN, CMID, CMID, 9, 1);

  // 2x criss-cross attention
  u16* cur = act1;
  u16* nxt[2] = {res1, res2};
  for (int r = 0; r < 2; ++r) {
    k_conv_gemm<<<dim3(MT, 1), 256, 0, stream>>>(
        cur, nullptr, CMID, wQ, nullptr, q_b, qb, nullptr, CMID, CQ, 64, 1, 0);
    k_conv_gemm<<<dim3(MT, 1), 256, 0, stream>>>(
        cur, nullptr, CMID, wK, nullptr, k_b, kb, nullptr, CMID, CQ, 64, 1, 0);
    k_conv_gemm<<<dim3(MT, 8), 256, 0, stream>>>(
        cur, nullptr, CMID, wV, nullptr, v_b, vbuf, nullptr, CMID, CMID, CMID, 1, 0);
    k_energy<<<dim3(BB * HH, 2), 224, 0, stream>>>(qb, kb, E);
    k_softmax<<<(PP + 255) / 256, 256, 0, stream>>>(E, attn);
    k_agg_row<<<dim3(BB * HH, 8), 256, 0, stream>>>(attn, vbuf, oatt);
    k_agg_col<<<dim3(BB * WWI, 8), 256, 0, stream>>>(attn, vbuf, oatt, cur, gamma, nxt[r]);
    cur = nxt[r];
  }

  // convb + bn2 + relu
  k_conv_gemm<<<dim3(MT, 8), 256, 0, stream>>>(
      cur, nullptr, CMID, wB, bn2i, bn2bt, act2, nullptr, CMID, CMID, CMID, 9, 1);
  // bottleneck conv on concat([x, out]) + bn3 + relu (two-source gather)
  k_conv_gemm<<<dim3(MT, 8), 256, 0, stream>>>(
      x_nhwc, act2, CIN, wBt, bn3i, bn3bt, ybuf, nullptr, CIN + CMID, CMID, CMID, 9, 1);
  // classifier 1x1 + bias -> NCHW f32 output
  k_conv_gemm<<<dim3(MT, 1), 256, 0, stream>>>(
      ybuf, nullptr, CMID, wCls, nullptr, cls_b, nullptr, (float*)d_out,
      CMID, NCLS, 64, 1, 0);
}
